// MultiHeadSelfAttention_15272903705113
// MI455X (gfx1250) — matrix operand order repaired and checked by reading, not compiled, not run on hardware
//
#include <hip/hip_runtime.h>

// ---------------- problem constants ----------------
#define Bsz 4
#define Tsz 2048
#define Dsz 1024
#define Hsz 16
#define HDsz 64
#define Msz (Bsz * Tsz)      // 8192 rows
#define NQKV (3 * Dsz)       // 3072
#define Ksz Dsz              // 1024

typedef __attribute__((ext_vector_type(16))) __bf16 v16bf;
typedef __attribute__((ext_vector_type(8)))  float  v8f;

union Frag { unsigned u[8]; v16bf v; };

// A-matrix (16-bit, 16x32) K index for VGPR v, lane-half h  (ISA 7.12.2)
__device__ __forceinline__ int kA(int v, int h) {
    return (v < 4 ? 2 * v : 16 + 2 * (v - 4)) + 8 * h;
}
// B-matrix (16-bit, 32x16) K index for VGPR v, lane-half h (same order as kA)
__device__ __forceinline__ int kB(int v, int h) {
    return (v < 4 ? 2 * v : 16 + 2 * (v - 4)) + 8 * h;
}

// LDS row stride: 40 elements = 80 bytes.
//  - 16-byte aligned rows -> fragment reads merge into ds_load_b128
//  - 20*r mod 64 distinct for r=0..15 -> conflict-free 16-row fragment reads
#define LDSS 40
#define ABUF (128 * LDSS)    // element offset of A double-buffer half
#define WBUF (64 * LDSS)     // element offset of W double-buffer half

// ---------------- fp32 -> bf16 conversion (vectorized) ----------------
__global__ __launch_bounds__(256) void cvt_bf16(const float* __restrict__ s,
                                                __bf16* __restrict__ d, int n4) {
    int i = blockIdx.x * 256 + threadIdx.x;
    if (i < n4) {
        float4 f = ((const float4*)s)[i];
        __bf16 o[4] = {(__bf16)f.x, (__bf16)f.y, (__bf16)f.z, (__bf16)f.w};
        *(uint2*)(d + (size_t)i * 4) = *(const uint2*)o;
    }
}

// ---------------- shared GEMM mainloop ----------------
// C[m][n] = sum_k A[m][k] * W[n][k]; block tile 128x64, wave tile 32x32.
// Double-buffered LDS with COMPILE-TIME buffer offsets (keeps addrspace(3)
// inference alive -> ds_load_b128/ds_store_b128, not flat).

__device__ __forceinline__ void g_load(const __bf16* __restrict__ Ablk,
                                       const __bf16* __restrict__ Wblk,
                                       int Kdim, int kk, int tid,
                                       uint4& a0, uint4& a1, uint4& w0) {
    int ar = tid >> 1, ac = (tid & 1) * 16;
    int aoff = ar * Kdim + kk + ac;                 // < 2^18, fits easily
    a0 = *(const uint4*)(Ablk + aoff);
    a1 = *(const uint4*)(Ablk + aoff + 8);
    int wr = tid >> 2, wc2 = (tid & 3) * 8;
    w0 = *(const uint4*)(Wblk + wr * Kdim + kk + wc2);
}

__device__ __forceinline__ void s_store(__bf16* As, __bf16* Ws,
                                        int aBufEl, int wBufEl, int tid,
                                        const uint4& a0, const uint4& a1,
                                        const uint4& w0) {
    int ar = tid >> 1, ac = (tid & 1) * 16;
    *(uint4*)(As + aBufEl + ar * LDSS + ac)     = a0;   // 16B aligned
    *(uint4*)(As + aBufEl + ar * LDSS + ac + 8) = a1;
    *(uint4*)(Ws + wBufEl + (tid >> 2) * LDSS + (tid & 3) * 8) = w0;
}

__device__ __forceinline__ void compute_step(const __bf16* As, const __bf16* Ws,
                                             int aBufEl, int wBufEl,
                                             int wr, int wc, int half, int ln,
                                             v8f (&c)[2][2]) {
    const unsigned* As32 = (const unsigned*)As;
    const unsigned* Ws32 = (const unsigned*)Ws;
    Frag af[2], bf[2];
#pragma unroll
    for (int s = 0; s < 2; ++s) {
        int arow = wr * 32 + s * 16 + ln;
#pragma unroll
        for (int v = 0; v < 8; ++v)
            af[s].u[v] = As32[(aBufEl + arow * LDSS + kA(v, half)) >> 1];
    }
#pragma unroll
    for (int nt = 0; nt < 2; ++nt) {
        int nrow = wc * 32 + nt * 16 + ln;
#pragma unroll
        for (int v = 0; v < 8; ++v)
            bf[nt].u[v] = Ws32[(wBufEl + nrow * LDSS + kB(v, half)) >> 1];
    }
#pragma unroll
    for (int s = 0; s < 2; ++s)
#pragma unroll
        for (int nt = 0; nt < 2; ++nt)
            c[s][nt] = __builtin_amdgcn_wmma_f32_16x16x32_bf16(
                false, af[s].v, false, bf[nt].v, (short)0, c[s][nt], false, false);
}

__device__ __forceinline__ void gemm_mainloop(const __bf16* __restrict__ Ablk,
                                              const __bf16* __restrict__ Wblk,
                                              int Kdim,
                                              __bf16* As, __bf16* Ws, int tid,
                                              v8f (&c)[2][2]) {
    const int wave = tid >> 5, lane = tid & 31;
    const int wr = wave & 3, wc = wave >> 2;   // 4 x 2 wave grid
    const int half = lane >> 4, ln = lane & 15;

    uint4 a0, a1, w0;
    g_load(Ablk, Wblk, Kdim, 0, tid, a0, a1, w0);
    s_store(As, Ws, 0, 0, tid, a0, a1, w0);

    // Kdim is a multiple of 64: unroll-by-2 makes buffer offsets literal.
    for (int kk = 0; kk < Kdim; kk += 64) {
        __syncthreads();                                    // buf0 stores visible
        g_load(Ablk, Wblk, Kdim, kk + 32, tid, a0, a1, w0); // always valid
        compute_step(As, Ws, 0, 0, wr, wc, half, ln, c);
        s_store(As, Ws, ABUF, WBUF, tid, a0, a1, w0);

        __syncthreads();                                    // buf1 stores visible
        bool more = (kk + 64) < Kdim;
        if (more) g_load(Ablk, Wblk, Kdim, kk + 64, tid, a0, a1, w0);
        compute_step(As, Ws, ABUF, WBUF, wr, wc, half, ln, c);
        if (more) s_store(As, Ws, 0, 0, tid, a0, a1, w0);
    }
}

// ---------------- QKV GEMM: qkv = x @ Wqkv^T, scatter to Q,K,Vt ----------------
__device__ __forceinline__ void store_qkv(float val, int n, int b, int t,
                                          __bf16* qb, __bf16* kb, __bf16* vtb) {
    int which = n >> 10;          // 0=q 1=k 2=v
    int c = n & 1023;
    int h = c >> 6, hd = c & 63;
    int bh = b * Hsz + h;
    __bf16 x = (__bf16)val;
    if (which == 0)      qb[((size_t)bh * Tsz + t) * HDsz + hd] = x;
    else if (which == 1) kb[((size_t)bh * Tsz + t) * HDsz + hd] = x;
    else                 vtb[((size_t)bh * HDsz + hd) * Tsz + t] = x;  // transposed
}

__global__ __launch_bounds__(256) void gemm_qkv(const __bf16* __restrict__ xb,
                                                const __bf16* __restrict__ wb,
                                                __bf16* __restrict__ qb,
                                                __bf16* __restrict__ kb,
                                                __bf16* __restrict__ vtb) {
    __shared__ __bf16 As[2 * 128 * LDSS];
    __shared__ __bf16 Ws[2 * 64 * LDSS];
    int tid = threadIdx.x;
    int wave = tid >> 5, lane = tid & 31;
    int wr = wave & 3, wc = wave >> 2;
    int half = lane >> 4, ln = lane & 15;
    int m0 = blockIdx.y * 128, n0 = blockIdx.x * 64;

    v8f c[2][2] = {{{}, {}}, {{}, {}}};
    gemm_mainloop(xb + (size_t)m0 * Ksz, wb + (size_t)n0 * Ksz, Ksz, As, Ws, tid, c);

#pragma unroll
    for (int s = 0; s < 2; ++s)
#pragma unroll
        for (int nt = 0; nt < 2; ++nt)
#pragma unroll
            for (int e = 0; e < 8; ++e) {
                int m = m0 + wr * 32 + s * 16 + e + half * 8;
                int b = m >> 11, t = m & 2047;
                int n = n0 + wc * 32 + nt * 16 + ln;
                store_qkv(c[s][nt][e], n, b, t, qb, kb, vtb);
            }
}

// ---------------- flash attention: 1 wave per 16-row Q tile ----------------
__global__ __launch_bounds__(128) void attn_kernel(const __bf16* __restrict__ qb,
                                                   const __bf16* __restrict__ kb,
                                                   const __bf16* __restrict__ vtb,
                                                   __bf16* __restrict__ attn) {
    __shared__ __bf16 ps[4][16 * LDSS];
    int tid = threadIdx.x;
    int wave = tid >> 5, lane = tid & 31;
    int half = lane >> 4, ln = lane & 15;
    int bh = blockIdx.x;
    int b = bh >> 4, h = bh & 15;
    int q0 = (blockIdx.y * 4 + wave) * 16;

    // block-uniform bases -> saddr global loads with 32-bit lane offsets
    const __bf16* kbase = kb + (size_t)bh * Tsz * HDsz;
    const __bf16* vbase = vtb + (size_t)bh * HDsz * Tsz;
    const unsigned* q32 = (const unsigned*)(qb + (size_t)bh * Tsz * HDsz);
    const unsigned* k32 = (const unsigned*)kbase;
    const unsigned* v32 = (const unsigned*)vbase;

    // Q A-fragments (hd 0..31 and 32..63), loaded once
    Frag qlo, qhi;
    int qoff = (q0 + ln) * HDsz;               // dwords below: (qoff + k)/2
#pragma unroll
    for (int v = 0; v < 8; ++v) {
        qlo.u[v] = q32[(qoff + kA(v, half)) >> 1];
        qhi.u[v] = q32[(qoff + 32 + kA(v, half)) >> 1];
    }

    float mrow[8], lrow[8];
    v8f o[4];
#pragma unroll
    for (int e = 0; e < 8; ++e) { mrow[e] = -3.0e38f; lrow[e] = 0.0f; }
#pragma unroll
    for (int nt = 0; nt < 4; ++nt) o[nt] = (v8f){};

    const float scale = 0.125f;  // 1/sqrt(64)

    for (int key0 = 0; key0 <= q0 + 15; key0 += 32) {
        // prefetch next chunk's K rows and V columns into cache (global_prefetch_b8)
        if (key0 + 32 <= q0 + 15) {
            __builtin_prefetch(kbase + (key0 + 32 + lane) * HDsz, 0, 1);
            __builtin_prefetch(vbase + lane * Tsz + key0 + 32, 0, 1);
        }

        // ---- S = scale * Q @ K^T, two 16-key column tiles ----
        v8f s0 = {}, s1 = {};
        {
            Frag blo, bhi;
            int kb0 = (key0 + ln) * HDsz;
#pragma unroll
            for (int v = 0; v < 8; ++v) {
                blo.u[v] = k32[(kb0 + kB(v, half)) >> 1];
                bhi.u[v] = k32[(kb0 + 32 + kB(v, half)) >> 1];
            }
            s0 = __builtin_amdgcn_wmma_f32_16x16x32_bf16(false, qlo.v, false, blo.v,
                                                         (short)0, s0, false, false);
            s0 = __builtin_amdgcn_wmma_f32_16x16x32_bf16(false, qhi.v, false, bhi.v,
                                                         (short)0, s0, false, false);
        }
        {
            Frag blo, bhi;
            int kb1 = (key0 + 16 + ln) * HDsz;
#pragma unroll
            for (int v = 0; v < 8; ++v) {
                blo.u[v] = k32[(kb1 + kB(v, half)) >> 1];
                bhi.u[v] = k32[(kb1 + 32 + kB(v, half)) >> 1];
            }
            s1 = __builtin_amdgcn_wmma_f32_16x16x32_bf16(false, qlo.v, false, blo.v,
                                                         (short)0, s1, false, false);
            s1 = __builtin_amdgcn_wmma_f32_16x16x32_bf16(false, qhi.v, false, bhi.v,
                                                         (short)0, s1, false, false);
        }

        // ---- online softmax (row reductions across 16-lane groups) ----
        bool needmask = (key0 + 31 > q0);
#pragma unroll
        for (int e = 0; e < 8; ++e) {
            float a0 = s0[e] * scale, a1 = s1[e] * scale;
            if (needmask) {
                int row = q0 + e + half * 8;
                if (key0 + ln > row)      a0 = -3.0e38f;
                if (key0 + 16 + ln > row) a1 = -3.0e38f;
            }
            float mx = fmaxf(a0, a1);
            mx = fmaxf(mx, __shfl_xor(mx, 1));
            mx = fmaxf(mx, __shfl_xor(mx, 2));
            mx = fmaxf(mx, __shfl_xor(mx, 4));
            mx = fmaxf(mx, __shfl_xor(mx, 8));
            float mnew = fmaxf(mrow[e], mx);
            float alpha = __expf(mrow[e] - mnew);
            float p0 = __expf(a0 - mnew);
            float p1 = __expf(a1 - mnew);
            float sum = p0 + p1;
            sum += __shfl_xor(sum, 1);
            sum += __shfl_xor(sum, 2);
            sum += __shfl_xor(sum, 4);
            sum += __shfl_xor(sum, 8);
            lrow[e] = lrow[e] * alpha + sum;
            mrow[e] = mnew;
            o[0][e] *= alpha; o[1][e] *= alpha; o[2][e] *= alpha; o[3][e] *= alpha;
            s0[e] = p0; s1[e] = p1;
        }

        // ---- transpose P (C-layout -> A-fragment) through per-wave LDS ----
        __bf16* pw = ps[wave];
#pragma unroll
        for (int e = 0; e < 8; ++e) {
            int row = e + half * 8;
            pw[row * LDSS + ln]      = (__bf16)s0[e];
            pw[row * LDSS + 16 + ln] = (__bf16)s1[e];
        }
        asm volatile("s_wait_dscnt 0" ::: "memory");  // DS in-order per wave; fence compiler too
        Frag pf;
        const unsigned* pw32 = (const unsigned*)pw;
#pragma unroll
        for (int v = 0; v < 8; ++v)
            pf.u[v] = pw32[(ln * LDSS + kA(v, half)) >> 1];

        // ---- O += P @ V  (V stored transposed [bh][hd][t] -> contiguous B frags) ----
#pragma unroll
        for (int nt = 0; nt < 4; ++nt) {
            Frag vf;
            int hd = nt * 16 + ln;
            int vo = hd * Tsz + key0;
#pragma unroll
            for (int v = 0; v < 8; ++v)
                vf.u[v] = v32[(vo + kB(v, half)) >> 1];
            o[nt] = __builtin_amdgcn_wmma_f32_16x16x32_bf16(false, pf.v, false, vf.v,
                                                            (short)0, o[nt], false, false);
        }
    }

    // ---- normalize + store attn rows [B*T, D] (head-concat layout) ----
    __bf16* abase = attn + ((size_t)b * Tsz) * Dsz + h * HDsz;   // uniform base
    float inv[8];
#pragma unroll
    for (int e = 0; e < 8; ++e) inv[e] = 1.0f / lrow[e];
#pragma unroll
    for (int nt = 0; nt < 4; ++nt) {
#pragma unroll
        for (int e = 0; e < 8; ++e) {
            int row = q0 + e + half * 8;
            int hd = nt * 16 + ln;
            abase[row * Dsz + hd] = (__bf16)(o[nt][e] * inv[e]);
        }
    }
}

// ---------------- projection GEMM: out = attn @ Wproj^T + b (fp32 out) ----------------
__global__ __launch_bounds__(256) void gemm_proj(const __bf16* __restrict__ ab,
                                                 const __bf16* __restrict__ wb,
                                                 const float* __restrict__ bias,
                                                 float* __restrict__ out) {
    __shared__ __bf16 As[2 * 128 * LDSS];
    __shared__ __bf16 Ws[2 * 64 * LDSS];
    int tid = threadIdx.x;
    int wave = tid >> 5, lane = tid & 31;
    int wr = wave & 3, wc = wave >> 2;
    int half = lane >> 4, ln = lane & 15;
    int m0 = blockIdx.y * 128, n0 = blockIdx.x * 64;

    v8f c[2][2] = {{{}, {}}, {{}, {}}};
    gemm_mainloop(ab + (size_t)m0 * Ksz, wb + (size_t)n0 * Ksz, Ksz, As, Ws, tid, c);

    float* obase = out + (size_t)m0 * Dsz + n0;    // uniform base
#pragma unroll
    for (int nt = 0; nt < 2; ++nt) {
        int n = wc * 32 + nt * 16 + ln;
        float bia = bias[n0 + n];
#pragma unroll
        for (int s = 0; s < 2; ++s)
#pragma unroll
            for (int e = 0; e < 8; ++e) {
                int m = wr * 32 + s * 16 + e + half * 8;
                obase[m * Dsz + n] = c[s][nt][e] + bia;
            }
    }
}

// ---------------- host launcher ----------------
extern "C" void kernel_launch(void* const* d_in, const int* in_sizes, int n_in,
                              void* d_out, int out_size, void* d_ws, size_t ws_size,
                              hipStream_t stream) {
    (void)in_sizes; (void)n_in; (void)out_size; (void)ws_size;
    const float* x     = (const float*)d_in[0];
    const float* Wqkv  = (const float*)d_in[1];
    const float* Wproj = (const float*)d_in[2];
    const float* bproj = (const float*)d_in[3];
    float* out = (float*)d_out;

    char* p = (char*)d_ws;
    size_t off = 0;
    auto alloc = [&](size_t bytes) -> void* {
        void* r = p + off;
        off += (bytes + 255) & ~(size_t)255;
        return r;
    };
    __bf16* xb    = (__bf16*)alloc((size_t)Msz * Ksz * 2);
    __bf16* wqkvb = (__bf16*)alloc((size_t)NQKV * Ksz * 2);
    __bf16* wprjb = (__bf16*)alloc((size_t)Dsz * Ksz * 2);
    __bf16* qb    = (__bf16*)alloc((size_t)Bsz * Hsz * Tsz * HDsz * 2);
    __bf16* kb    = (__bf16*)alloc((size_t)Bsz * Hsz * Tsz * HDsz * 2);
    __bf16* vtb   = (__bf16*)alloc((size_t)Bsz * Hsz * Tsz * HDsz * 2);
    __bf16* attnb = (__bf16*)alloc((size_t)Msz * Dsz * 2);

    int nx4 = Msz * Ksz / 4, nwq4 = NQKV * Ksz / 4, nwp4 = Dsz * Ksz / 4;
    cvt_bf16<<<(nx4 + 255) / 256, 256, 0, stream>>>(x, xb, nx4);
    cvt_bf16<<<(nwq4 + 255) / 256, 256, 0, stream>>>(Wqkv, wqkvb, nwq4);
    cvt_bf16<<<(nwp4 + 255) / 256, 256, 0, stream>>>(Wproj, wprjb, nwp4);

    gemm_qkv<<<dim3(NQKV / 64, Msz / 128), 256, 0, stream>>>(xb, wqkvb, qb, kb, vtb);

    attn_kernel<<<dim3(Bsz * Hsz, Tsz / 64), 128, 0, stream>>>(qb, kb, vtb, attnb);

    gemm_proj<<<dim3(Dsz / 64, Msz / 128), 256, 0, stream>>>(attnb, wprjb, bproj, out);
}
